// PrRoIPool2D_9955734192457
// MI455X (gfx1250) — matrix operand order repaired and checked by reading, not compiled, not run on hardware
//
#include <hip/hip_runtime.h>

typedef _Float16 v16h __attribute__((ext_vector_type(16)));
typedef float    v8f  __attribute__((ext_vector_type(8)));

#define PHB 7
#define PWB 7
#define SCALE 0.0625f
#define CC 256
#define HDIM 152
#define WDIM 152
#define HWSZ (HDIM*WDIM)
#define CHW (CC*HWSZ)
#define MAXRB 32     // max rois staged per workspace batch
#define MAXSTEP 72   // max K-steps (window <= 48x48 -> 2304/32)

__device__ __forceinline__ float hat_int(float t) {
  // antiderivative of hat(t)=max(0,1-|t|), clamped to [0,1]
  t = fminf(1.f, fmaxf(-1.f, t));
  float a = t + 1.f, b = 1.f - t;
  return (t <= 0.f) ? 0.5f * a * a : 1.f - 0.5f * b * b;
}

__device__ __forceinline__ unsigned short f2h_bits(float f) {
  union { _Float16 h; unsigned short u; } x;
  x.h = (_Float16)f;
  return x.u;
}

// ---------------------------------------------------------------------------
// Fallback: exact scalar PrRoI pooling (runs only if workspace is too small).
// ---------------------------------------------------------------------------
__global__ __launch_bounds__(64)
void prroi_scalar_kernel(const float* __restrict__ feat,
                         const float* __restrict__ rois,
                         float* __restrict__ out) {
  const int c   = blockIdx.x;
  const int r   = blockIdx.y;
  const int tid = threadIdx.x;
  if (tid >= 49) return;
  const int p = tid / 7;
  const int q = tid - p * 7;

  const float* roi = rois + r * 5;
  const int   bidx = (int)roi[0];
  const float x1 = roi[1] * SCALE, y1 = roi[2] * SCALE;
  const float x2 = roi[3] * SCALE, y2 = roi[4] * SCALE;
  const float bw = fmaxf(x2 - x1, 0.f) * (1.f / PWB);
  const float bh = fmaxf(y2 - y1, 0.f) * (1.f / PHB);
  const float area = bw * bh;
  const float inv_area = (area > 0.f) ? 1.f / fmaxf(area, 1e-12f) : 0.f;

  const float xlo = x1 + q * bw, xhi = xlo + bw;
  const float ylo = y1 + p * bh, yhi = ylo + bh;

  int w0 = min(WDIM - 1, max(0, (int)floorf(x1)));
  int w1 = max(w0, min(WDIM - 1, (int)ceilf(x2)));
  int h0 = min(HDIM - 1, max(0, (int)floorf(y1)));
  int h1 = max(h0, min(HDIM - 1, (int)ceilf(y2)));

  const float* f = feat + ((size_t)bidx * CC + c) * HWSZ;

  float sum = 0.f;
  for (int h = h0; h <= h1; ++h) {
    float wy = hat_int(yhi - (float)h) - hat_int(ylo - (float)h);
    if (wy != 0.f) {
      const float* fr = f + (size_t)h * WDIM;
      float row = 0.f;
      for (int w = w0; w <= w1; ++w) {
        float wx = hat_int(xhi - (float)w) - hat_int(xlo - (float)w);
        row += wx * fr[w];
      }
      sum += wy * row;
    }
  }
  out[((size_t)r * CC + c) * 49 + tid] = sum * inv_area;
}

// ---------------------------------------------------------------------------
// Stage A/B WMMA fragments (fp16) to workspace in contiguous-k (k = 16h + e) fragment
// order so the GEMM kernel uses contiguous v16h global loads.
// A frag: M = lane%16 (channel); K = e (+16 for upper half-wave).
// B frag: col n = lane%16 ; K = e (+16 for upper half-wave).
// ---------------------------------------------------------------------------
__global__ __launch_bounds__(256)
void prroi_stage_kernel(const float* __restrict__ feat,
                        const float* __restrict__ rois, int r0,
                        unsigned short* __restrict__ Afr,
                        unsigned short* __restrict__ Bfr,
                        int* __restrict__ hks, float* __restrict__ hinv) {
  const int rb  = blockIdx.x;
  const int r   = r0 + rb;
  const int tid = threadIdx.x;

  const float* roi = rois + r * 5;
  const int   bidx = (int)roi[0];
  const float x1 = roi[1] * SCALE, y1 = roi[2] * SCALE;
  const float x2 = roi[3] * SCALE, y2 = roi[4] * SCALE;
  const float bw = fmaxf(x2 - x1, 0.f) * (1.f / PWB);
  const float bh = fmaxf(y2 - y1, 0.f) * (1.f / PHB);
  const float area = bw * bh;
  const float inv_area = (area > 0.f) ? 1.f / fmaxf(area, 1e-12f) : 0.f;

  int w0 = min(WDIM - 1, max(0, (int)floorf(x1)));
  int w1 = max(w0, min(WDIM - 1, (int)ceilf(x2)));
  int nw = min(48, w1 - w0 + 1);
  int h0 = min(HDIM - 1, max(0, (int)floorf(y1)));
  int h1 = max(h0, min(HDIM - 1, (int)ceilf(y2)));
  int nh = min(48, h1 - h0 + 1);
  int Ktot   = nh * nw;
  int ksteps = (Ktot + 31) >> 5;             // <= MAXSTEP

  if (tid == 0) { hks[rb] = ksteps; hinv[rb] = inv_area; }

  const float* f = feat + (size_t)bidx * CHW;

  // A fragments: decode window cell once, reuse across the 16 channel tiles.
  for (int idx = tid; idx < ksteps * 512; idx += 256) {
    int s  = idx >> 9;
    int le = idx & 511;
    int l  = le >> 4, e = le & 15;
    int kl = e + ((l < 16) ? 0 : 16);
    int kg = s * 32 + kl;
    bool kin = kg < Ktot;
    size_t foff = 0;
    if (kin) {
      int hh = kg / nw, ww = kg - hh * nw;
      foff = (size_t)(h0 + hh) * WDIM + (w0 + ww);
    }
    const float* fp = f + (size_t)(l & 15) * HWSZ + foff;
    #pragma unroll 1
    for (int mtile = 0; mtile < 16; ++mtile) {
      float v = kin ? fp[(size_t)mtile * 16 * HWSZ] : 0.f;
      Afr[((size_t)(rb * 16 + mtile) * MAXSTEP) * 512 + idx] = f2h_bits(v);
    }
  }
  // B fragments: B[k, n] = Wy[p, hh] * Wx[q, ww]; decode once, 4 bin tiles.
  for (int idx = tid; idx < ksteps * 512; idx += 256) {
    int s  = idx >> 9;
    int le = idx & 511;
    int l  = le >> 4, e = le & 15;
    int kl = e + ((l < 16) ? 0 : 16);
    int kg = s * 32 + kl;
    bool kin = kg < Ktot;
    int hh = 0, ww = 0;
    if (kin) { hh = kg / nw; ww = kg - hh * nw; }
    float gx = (float)(w0 + ww), gy = (float)(h0 + hh);
    int nb = l & 15;
    #pragma unroll 1
    for (int nt = 0; nt < 4; ++nt) {
      int n = nt * 16 + nb;
      float v = 0.f;
      if (kin && n < 49) {
        int pp = n / 7, qq = n - pp * 7;
        float xl = x1 + qq * bw, yl = y1 + pp * bh;
        float wxv = hat_int(xl + bw - gx) - hat_int(xl - gx);
        float wyv = hat_int(yl + bh - gy) - hat_int(yl - gy);
        v = wxv * wyv;
      }
      Bfr[((size_t)(rb * 4 + nt) * MAXSTEP) * 512 + idx] = f2h_bits(v);
    }
  }
}

// ---------------------------------------------------------------------------
// WMMA GEMM: one wave per (roi, mtile, ntile). Unroll-by-2 K-loop with dual
// accumulators (two independent WMMA chains) + uniform-branch tail.
// Operands are direct v16h global loads (the toolchain-proven pattern).
// ---------------------------------------------------------------------------
__global__ __launch_bounds__(32)
void prroi_gemm_kernel(const unsigned short* __restrict__ Afr,
                       const unsigned short* __restrict__ Bfr,
                       const int* __restrict__ hks,
                       const float* __restrict__ hinv,
                       int r0, float* __restrict__ out) {
  const int rb    = blockIdx.x;        // roi within batch
  const int tile  = blockIdx.y;        // 64 = 16 mtiles x 4 ntiles
  const int mtile = tile >> 2;
  const int nt    = tile & 3;
  const int r     = r0 + rb;
  const int lane  = threadIdx.x;

  const int   ksteps   = hks[rb];      // wave-uniform
  const float inv_area = hinv[rb];

  const _Float16* Ab = (const _Float16*)Afr
      + ((size_t)(rb * 16 + mtile) * MAXSTEP) * 512 + lane * 16;
  const _Float16* Bb = (const _Float16*)Bfr
      + ((size_t)(rb * 4 + nt) * MAXSTEP) * 512 + lane * 16;

  v8f c0 = {}, c1 = {};
  int s = 0;
  #pragma unroll 1
  for (; s + 2 <= ksteps; s += 2) {
    v16h a0 = *(const v16h*)(Ab + (size_t)s * 512);
    v16h b0 = *(const v16h*)(Bb + (size_t)s * 512);
    v16h a1 = *(const v16h*)(Ab + (size_t)(s + 1) * 512);
    v16h b1 = *(const v16h*)(Bb + (size_t)(s + 1) * 512);
    c0 = __builtin_amdgcn_wmma_f32_16x16x32_f16(
        false, a0, false, b0, (short)0, c0, false, false);
    c1 = __builtin_amdgcn_wmma_f32_16x16x32_f16(
        false, a1, false, b1, (short)0, c1, false, false);
  }
  if (s < ksteps) {                    // uniform branch: EXEC stays all-ones
    v16h a = *(const v16h*)(Ab + (size_t)s * 512);
    v16h b = *(const v16h*)(Bb + (size_t)s * 512);
    c0 = __builtin_amdgcn_wmma_f32_16x16x32_f16(
        false, a, false, b, (short)0, c0, false, false);
  }
  v8f c = c0 + c1;

  // D: lane gives N = lane%16; VGPR v gives M = v (+8 for upper half-wave)
  const int n = nt * 16 + (lane & 15);
  if (n < 49) {
    size_t base = ((size_t)r * CC + mtile * 16) * 49 + n;
    int mo = (lane < 16) ? 0 : 8;
    #pragma unroll
    for (int v = 0; v < 8; v++)
      out[base + (size_t)(mo + v) * 49] = c[v] * inv_area;
  }
}

extern "C" void kernel_launch(void* const* d_in, const int* in_sizes, int n_in,
                              void* d_out, int out_size, void* d_ws, size_t ws_size,
                              hipStream_t stream) {
  const float* feat = (const float*)d_in[0];
  const float* rois = (const float*)d_in[1];
  float* out = (float*)d_out;
  const int R = in_sizes[1] / 5;       // 512 rois

  // Workspace sizing: ~1.41 MB per staged roi (A: 16 mtiles, B: 4 ntiles).
  const size_t perA = (size_t)16 * MAXSTEP * 512 * 2;
  const size_t perB = (size_t)4  * MAXSTEP * 512 * 2;
  int rbatch = 0;
  if (ws_size > 4096)
    rbatch = (int)((ws_size - 4096) / (perA + perB));
  if (rbatch > MAXRB) rbatch = MAXRB;

  if (rbatch < 1) {
    // Tiny workspace: exact scalar fallback.
    dim3 gscalar(CC, R);
    prroi_scalar_kernel<<<gscalar, 64, 0, stream>>>(feat, rois, out);
    return;
  }

  int*   hks  = (int*)d_ws;
  float* hinv = (float*)((char*)d_ws + 2048);
  unsigned short* Afr = (unsigned short*)((char*)d_ws + 4096);
  unsigned short* Bfr = (unsigned short*)((char*)d_ws + 4096 + perA * rbatch);

  for (int r0 = 0; r0 < R; r0 += rbatch) {
    int nb = (R - r0 < rbatch) ? (R - r0) : rbatch;
    prroi_stage_kernel<<<dim3(nb), 256, 0, stream>>>(
        feat, rois, r0, Afr, Bfr, hks, hinv);
    prroi_gemm_kernel<<<dim3(nb, 64), 32, 0, stream>>>(
        Afr, Bfr, hks, hinv, r0, out);
  }
}
